// AttentionGNN_28441273434168
// MI455X (gfx1250) — matrix operand order repaired and checked by reading, not compiled, not run on hardware
//
#include <hip/hip_runtime.h>

// ---------------------------------------------------------------------------
// Problem constants (from reference)
// ---------------------------------------------------------------------------
#define NN      50000
#define EE      800000
#define ET      (EE + NN)      // edges incl. self loops
#define GG      500
#define F_IN    74
#define HDIM    256
#define LL      4
#define NHEAD   8
#define HD      32
#define SLOPE   0.2f

typedef __attribute__((ext_vector_type(16))) __bf16 v16bf;
typedef __attribute__((ext_vector_type(8)))  __bf16 v8bf;
typedef __attribute__((ext_vector_type(8)))  float  v8f;

// ---------------------------------------------------------------------------
// Helpers
// ---------------------------------------------------------------------------
__device__ __forceinline__ unsigned short f32_to_bf16_rne(float f) {
    unsigned int u = __float_as_uint(f);
    unsigned int r = 0x7FFFu + ((u >> 16) & 1u);
    return (unsigned short)((u + r) >> 16);
}
__device__ __forceinline__ unsigned int encOrd(float f) {   // order-preserving f32->u32
    unsigned int u = __float_as_uint(f);
    return (u & 0x80000000u) ? ~u : (u | 0x80000000u);
}
__device__ __forceinline__ float decOrd(unsigned int x) {
    return (x & 0x80000000u) ? __uint_as_float(x & 0x7FFFFFFFu) : __uint_as_float(~x);
}

// ---------------------------------------------------------------------------
// Generic utility kernels
// ---------------------------------------------------------------------------
__global__ void zero_kernel(float* p, size_t n) {
    size_t t = (size_t)blockIdx.x * blockDim.x + threadIdx.x;
    if (t < n) p[t] = 0.0f;
}

// W[k][n] (256x256 f32, row major) -> Wt[n][k] (bf16)
__global__ __launch_bounds__(256)
void transpose_bf16_kernel(const float* __restrict__ W, unsigned short* __restrict__ Wt) {
    int n = blockIdx.x, k = threadIdx.x;
    Wt[n * HDIM + k] = f32_to_bf16_rne(W[k * HDIM + n]);
}

// ---------------------------------------------------------------------------
// WMMA GEMM:  C[M x 256] = A[M x 256](bf16) * W + bias, with Bt[n][k] = W[k][n]
// block = 256 threads (8 waves). Each wave owns a 16x32 output tile
// (two 16x16 accumulators sharing one A fragment -> 2 WMMA per A load).
// Block tile = 16 rows x 256 cols; A tile (8 KB) staged in LDS, shared 8x.
// grid = M/16 (M = 50000 divides exactly). EXEC uniform everywhere.
// ---------------------------------------------------------------------------
__global__ __launch_bounds__(256)
void gemm_wmma_256(const unsigned short* __restrict__ A,
                   const unsigned short* __restrict__ Bt,
                   const float* __restrict__ bias,
                   float* __restrict__ C, int M) {
    __shared__ __align__(16) unsigned short sA[16 * HDIM];   // 8 KB
    const int mBase = blockIdx.x * 16;
    const int tid   = threadIdx.x;

    // cooperative stage of A tile (16 x 256 bf16 = 512 uint4)
    {
        const uint4* s4 = (const uint4*)(A + (size_t)mBase * HDIM);
        uint4* d4 = (uint4*)sA;
        d4[tid]       = s4[tid];
        d4[tid + 256] = s4[tid + 256];
    }
    __syncthreads();

    const int wave = tid >> 5;
    const int lane = tid & 31;
    const int half = lane >> 4;                 // 0 or 1
    const int colL0 = wave * 32 + (lane & 15);  // first 16x16 tile column
    const int colL1 = colL0 + 16;               // second 16x16 tile column
    const unsigned short* brow0 = Bt + (size_t)colL0 * HDIM;
    const unsigned short* brow1 = Bt + (size_t)colL1 * HDIM;

    v8f acc0 = {}, acc1 = {};
    #pragma unroll
    for (int kb = 0; kb < HDIM; kb += 32) {
        // A fragment: lane holds row (lane&15), K = kb + half*16 + {0..15}
        const unsigned short* ap = sA + (lane & 15) * HDIM + kb + half * 16;
        v8bf alo = *(const v8bf*)ap;
        v8bf ahi = *(const v8bf*)(ap + 8);
        v16bf a;
        #pragma unroll
        for (int i = 0; i < 8; ++i) { a[i] = alo[i]; a[i + 8] = ahi[i]; }
        // B fragments: lane holds one column, 16 consecutive K each
        v16bf b0 = *(const v16bf*)(brow0 + kb + half * 16);
        v16bf b1 = *(const v16bf*)(brow1 + kb + half * 16);
        acc0 = __builtin_amdgcn_wmma_f32_16x16x32_bf16(false, a, false, b0,
                                                       (short)0, acc0, false, false);
        acc1 = __builtin_amdgcn_wmma_f32_16x16x32_bf16(false, a, false, b1,
                                                       (short)0, acc1, false, false);
    }
    const float bv0 = bias[colL0];
    const float bv1 = bias[colL1];
    #pragma unroll
    for (int r = 0; r < 8; ++r) {
        int m = mBase + r + half * 8;
        C[(size_t)m * HDIM + colL0] = acc0[r] + bv0;
        C[(size_t)m * HDIM + colL1] = acc1[r] + bv1;
    }
}

// ---------------------------------------------------------------------------
// Input linear: h = x @ in_w + in_b  (K = 74, VALU) — also emits bf16 copy
// ---------------------------------------------------------------------------
__global__ __launch_bounds__(256)
void input_linear_kernel(const float* __restrict__ x, const float* __restrict__ w,
                         const float* __restrict__ b, float* __restrict__ h,
                         unsigned short* __restrict__ hbf) {
    __shared__ float sx[F_IN];
    const int node = blockIdx.x, j = threadIdx.x;
    if (j < F_IN) sx[j] = x[(size_t)node * F_IN + j];
    __syncthreads();
    float acc = b[j];
    #pragma unroll 2
    for (int k = 0; k < F_IN; ++k) acc = fmaf(sx[k], w[k * HDIM + j], acc);
    const size_t idx = (size_t)node * HDIM + j;
    h[idx]   = acc;
    hbf[idx] = f32_to_bf16_rne(acc);
}

// ---------------------------------------------------------------------------
// GATv2 edge kernels — one thread per (edge, head)
// ---------------------------------------------------------------------------
__global__ void edge_score_kernel(const float* __restrict__ xl, const float* __restrict__ xr,
                                  const float* __restrict__ att, const int* __restrict__ ei,
                                  float* __restrict__ score, unsigned int* __restrict__ smaxU) {
    int t = blockIdx.x * blockDim.x + threadIdx.x;
    if (t >= ET * NHEAD) return;
    int e = t >> 3, h = t & 7;
    int s = (e < EE) ? ei[e]      : (e - EE);
    int d = (e < EE) ? ei[EE + e] : (e - EE);
    const float* pl = xl + (size_t)s * HDIM + h * HD;
    const float* pr = xr + (size_t)d * HDIM + h * HD;
    const float* pa = att + h * HD;
    float sc = 0.0f;
    #pragma unroll
    for (int k = 0; k < HD; ++k) {
        float v = pl[k] + pr[k];
        v = (v > 0.0f) ? v : SLOPE * v;
        sc = fmaf(v, pa[k], sc);
    }
    score[t] = sc;
    atomicMax(&smaxU[d * NHEAD + h], encOrd(sc));
}

__global__ void edge_num_kernel(const float* __restrict__ score,
                                const unsigned int* __restrict__ smaxU,
                                const int* __restrict__ ei,
                                float* __restrict__ num, float* __restrict__ den) {
    int t = blockIdx.x * blockDim.x + threadIdx.x;
    if (t >= ET * NHEAD) return;
    int e = t >> 3, h = t & 7;
    int d = (e < EE) ? ei[EE + e] : (e - EE);
    float v = __expf(score[t] - decOrd(smaxU[d * NHEAD + h]));
    num[t] = v;
    atomicAdd(&den[d * NHEAD + h], v);
}

__global__ void edge_agg_kernel(const float* __restrict__ xl, const float* __restrict__ num,
                                const float* __restrict__ den, const int* __restrict__ ei,
                                float* __restrict__ gout) {
    int t = blockIdx.x * blockDim.x + threadIdx.x;
    if (t >= ET * NHEAD) return;
    int e = t >> 3, h = t & 7;
    int s = (e < EE) ? ei[e]      : (e - EE);
    int d = (e < EE) ? ei[EE + e] : (e - EE);
    float alpha = num[t] / (den[d * NHEAD + h] + 1e-16f);
    const float* pl = xl + (size_t)s * HDIM + h * HD;
    float* po = gout + (size_t)d * HDIM + h * HD;
    #pragma unroll
    for (int k = 0; k < HD; ++k) atomicAdd(&po[k], pl[k] * alpha);
}

// ---------------------------------------------------------------------------
// bias (+residual) + LayerNorm + ReLU, one block (256 thr) per node.
// Writes fp32 h (residual for next layer) and bf16 copy (GEMM input).
// ---------------------------------------------------------------------------
__global__ __launch_bounds__(256)
void post_layer_kernel(const float* __restrict__ gout, const float* __restrict__ res,
                       const float* __restrict__ gat_b, const float* __restrict__ ln_g,
                       const float* __restrict__ ln_b, int addRes,
                       float* __restrict__ hOut, unsigned short* __restrict__ hbf) {
    __shared__ float red[256];
    const int n = blockIdx.x, j = threadIdx.x;
    const size_t idx = (size_t)n * HDIM + j;
    float v = gout[idx] + gat_b[j];
    if (addRes) v += res[idx];
    red[j] = v; __syncthreads();
    for (int s = 128; s > 0; s >>= 1) { if (j < s) red[j] += red[j + s]; __syncthreads(); }
    float mean = red[0] * (1.0f / HDIM);
    __syncthreads();
    float dv = v - mean;
    red[j] = dv * dv; __syncthreads();
    for (int s = 128; s > 0; s >>= 1) { if (j < s) red[j] += red[j + s]; __syncthreads(); }
    float var = red[0] * (1.0f / HDIM);
    float y = dv * rsqrtf(var + 1e-5f) * ln_g[j] + ln_b[j];
    y = (y > 0.0f) ? y : 0.0f;
    hOut[idx] = y;
    hbf[idx]  = f32_to_bf16_rne(y);
}

// ---------------------------------------------------------------------------
// Pooling + MLP tail
// ---------------------------------------------------------------------------
__global__ void pool_accum_kernel(const float* __restrict__ ho, const int* __restrict__ batch,
                                  float* __restrict__ gr) {
    size_t t = (size_t)blockIdx.x * blockDim.x + threadIdx.x;
    if (t >= (size_t)NN * HDIM) return;
    int n = (int)(t >> 8), j = (int)(t & 255);
    atomicAdd(&gr[(size_t)batch[n] * HDIM + j], ho[t]);
}
__global__ void pool_count_kernel(const int* __restrict__ batch, float* __restrict__ cnt) {
    int n = blockIdx.x * blockDim.x + threadIdx.x;
    if (n < NN) atomicAdd(&cnt[batch[n]], 1.0f);
}
__global__ void pool_div_kernel(float* __restrict__ gr, const float* __restrict__ cnt) {
    int t = blockIdx.x * blockDim.x + threadIdx.x;
    if (t >= GG * HDIM) return;
    gr[t] /= fmaxf(cnt[t >> 8], 1.0f);
}

__global__ void linear_kernel(const float* __restrict__ A, const float* __restrict__ W,
                              const float* __restrict__ b, float* __restrict__ C,
                              int M, int K, int Nc) {
    int t = blockIdx.x * blockDim.x + threadIdx.x;
    if (t >= M * Nc) return;
    int m = t / Nc, j = t % Nc;
    float acc = b[j];
    for (int k = 0; k < K; ++k) acc = fmaf(A[m * K + k], W[k * Nc + j], acc);
    C[t] = acc;
}

// one block (256 thr) per column
__global__ __launch_bounds__(256)
void batchnorm_relu_kernel(const float* __restrict__ Z, const float* __restrict__ g,
                           const float* __restrict__ b, float* __restrict__ Zn,
                           int M, int Nc) {
    __shared__ float rs[256], rs2[256];
    int j = blockIdx.x, tid = threadIdx.x;
    float s = 0.0f, s2 = 0.0f;
    for (int m = tid; m < M; m += 256) { float v = Z[m * Nc + j]; s += v; s2 += v * v; }
    rs[tid] = s; rs2[tid] = s2; __syncthreads();
    for (int st = 128; st > 0; st >>= 1) {
        if (tid < st) { rs[tid] += rs[tid + st]; rs2[tid] += rs2[tid + st]; }
        __syncthreads();
    }
    float mean = rs[0] / M;
    float var  = rs2[0] / M - mean * mean;
    float inv  = rsqrtf(var + 1e-5f);
    for (int m = tid; m < M; m += 256) {
        float v = (Z[m * Nc + j] - mean) * inv * g[j] + b[j];
        Zn[m * Nc + j] = (v > 0.0f) ? v : 0.0f;
    }
}

// ---------------------------------------------------------------------------
// Launch
// ---------------------------------------------------------------------------
extern "C" void kernel_launch(void* const* d_in, const int* in_sizes, int n_in,
                              void* d_out, int out_size, void* d_ws, size_t ws_size,
                              hipStream_t stream) {
    (void)in_sizes; (void)n_in; (void)out_size; (void)ws_size;
    const float* x     = (const float*)d_in[0];
    const int*   ei    = (const int*)  d_in[1];
    const int*   batch = (const int*)  d_in[2];
    const float* in_w  = (const float*)d_in[3];
    const float* in_b  = (const float*)d_in[4];
    const float* wl    = (const float*)d_in[5];
    const float* wl_b  = (const float*)d_in[6];
    const float* wr    = (const float*)d_in[7];
    const float* wr_b  = (const float*)d_in[8];
    const float* att   = (const float*)d_in[9];
    const float* gat_b = (const float*)d_in[10];
    const float* ln_g  = (const float*)d_in[11];
    const float* ln_b  = (const float*)d_in[12];
    const float* out_w = (const float*)d_in[13];
    const float* out_b = (const float*)d_in[14];
    const float* p1_w  = (const float*)d_in[15];
    const float* p1_b  = (const float*)d_in[16];
    const float* bn1_g = (const float*)d_in[17];
    const float* bn1_b = (const float*)d_in[18];
    const float* p2_w  = (const float*)d_in[19];
    const float* p2_b  = (const float*)d_in[20];
    const float* bn2_g = (const float*)d_in[21];
    const float* bn2_b = (const float*)d_in[22];
    const float* p3_w  = (const float*)d_in[23];
    const float* p3_b  = (const float*)d_in[24];
    float* out = (float*)d_out;

    // ---- carve workspace ----
    char* wptr = (char*)d_ws;
    auto carve = [&](size_t bytes) { void* p = wptr; wptr += (bytes + 255) & ~(size_t)255; return p; };
    float*          hA    = (float*)carve((size_t)NN * HDIM * 4);
    float*          hB    = (float*)carve((size_t)NN * HDIM * 4);
    unsigned short* hbf   = (unsigned short*)carve((size_t)NN * HDIM * 2);
    float*          xl    = (float*)carve((size_t)NN * HDIM * 4);
    float*          xr    = (float*)carve((size_t)NN * HDIM * 4);
    float*          gout  = (float*)carve((size_t)NN * HDIM * 4);
    float*          score = (float*)carve((size_t)ET * NHEAD * 4);
    float*          num   = (float*)carve((size_t)ET * NHEAD * 4);
    unsigned int*   smaxU = (unsigned int*)carve((size_t)NN * NHEAD * 4);
    float*          den   = (float*)carve((size_t)NN * NHEAD * 4);
    unsigned short* wlT   = (unsigned short*)carve((size_t)LL * HDIM * HDIM * 2);
    unsigned short* wrT   = (unsigned short*)carve((size_t)LL * HDIM * HDIM * 2);
    unsigned short* owT   = (unsigned short*)carve((size_t)HDIM * HDIM * 2);
    float*          gr    = (float*)carve((size_t)GG * HDIM * 4);
    float*          cnt   = (float*)carve((size_t)GG * 4);
    float*          z1    = (float*)carve((size_t)GG * 128 * 4);
    float*          z1n   = (float*)carve((size_t)GG * 128 * 4);
    float*          z2    = (float*)carve((size_t)GG * 64 * 4);
    float*          z2n   = (float*)carve((size_t)GG * 64 * 4);

    auto cdiv = [](size_t a, size_t b) { return (unsigned)((a + b - 1) / b); };

    // ---- weight prep: bf16 transpose (L2 resident) ----
    for (int i = 0; i < LL; ++i) {
        transpose_bf16_kernel<<<HDIM, HDIM, 0, stream>>>(wl + (size_t)i * HDIM * HDIM,
                                                         wlT + (size_t)i * HDIM * HDIM);
        transpose_bf16_kernel<<<HDIM, HDIM, 0, stream>>>(wr + (size_t)i * HDIM * HDIM,
                                                         wrT + (size_t)i * HDIM * HDIM);
    }
    transpose_bf16_kernel<<<HDIM, HDIM, 0, stream>>>(out_w, owT);

    // ---- input projection (fp32 + bf16 epilogue) ----
    input_linear_kernel<<<NN, HDIM, 0, stream>>>(x, in_w, in_b, hA, hbf);

    float* h_cur = hA;
    float* h_next = hB;
    const size_t nfeat = (size_t)NN * HDIM;
    const size_t nhead_tot = (size_t)ET * NHEAD;
    const unsigned ggrid = NN / 16;   // 3125 blocks, 16 rows x 256 cols each

    for (int i = 0; i < LL; ++i) {
        zero_kernel<<<cdiv(nfeat, 256), 256, 0, stream>>>(gout, nfeat);
        zero_kernel<<<cdiv((size_t)NN * NHEAD, 256), 256, 0, stream>>>((float*)smaxU, (size_t)NN * NHEAD);
        zero_kernel<<<cdiv((size_t)NN * NHEAD, 256), 256, 0, stream>>>(den, (size_t)NN * NHEAD);

        gemm_wmma_256<<<ggrid, 256, 0, stream>>>(hbf, wlT + (size_t)i * HDIM * HDIM,
                                                 wl_b + i * HDIM, xl, NN);
        gemm_wmma_256<<<ggrid, 256, 0, stream>>>(hbf, wrT + (size_t)i * HDIM * HDIM,
                                                 wr_b + i * HDIM, xr, NN);

        edge_score_kernel<<<cdiv(nhead_tot, 256), 256, 0, stream>>>(xl, xr, att + i * HDIM,
                                                                    ei, score, smaxU);
        edge_num_kernel<<<cdiv(nhead_tot, 256), 256, 0, stream>>>(score, smaxU, ei, num, den);
        edge_agg_kernel<<<cdiv(nhead_tot, 256), 256, 0, stream>>>(xl, num, den, ei, gout);

        post_layer_kernel<<<NN, HDIM, 0, stream>>>(gout, h_cur, gat_b + i * HDIM,
                                                   ln_g + i * HDIM, ln_b + i * HDIM,
                                                   (i > 0) ? 1 : 0, h_next, hbf);
        float* tmp = h_cur; h_cur = h_next; h_next = tmp;
    }

    // ---- output projection (reuse xl as ho; hbf holds last layer's bf16) ----
    gemm_wmma_256<<<ggrid, 256, 0, stream>>>(hbf, owT, out_b, xl, NN);

    // ---- mean pooling per graph ----
    zero_kernel<<<cdiv((size_t)GG * HDIM, 256), 256, 0, stream>>>(gr, (size_t)GG * HDIM);
    zero_kernel<<<cdiv((size_t)GG, 256), 256, 0, stream>>>(cnt, (size_t)GG);
    pool_accum_kernel<<<cdiv(nfeat, 256), 256, 0, stream>>>(xl, batch, gr);
    pool_count_kernel<<<cdiv((size_t)NN, 256), 256, 0, stream>>>(batch, cnt);
    pool_div_kernel<<<cdiv((size_t)GG * HDIM, 256), 256, 0, stream>>>(gr, cnt);

    // ---- MLP head ----
    linear_kernel<<<cdiv((size_t)GG * 128, 256), 256, 0, stream>>>(gr, p1_w, p1_b, z1, GG, HDIM, 128);
    batchnorm_relu_kernel<<<128, 256, 0, stream>>>(z1, bn1_g, bn1_b, z1n, GG, 128);
    linear_kernel<<<cdiv((size_t)GG * 64, 256), 256, 0, stream>>>(z1n, p2_w, p2_b, z2, GG, 128, 64);
    batchnorm_relu_kernel<<<64, 256, 0, stream>>>(z2, bn2_g, bn2_b, z2n, GG, 64);
    linear_kernel<<<cdiv((size_t)GG, 256), 256, 0, stream>>>(z2n, p3_w, p3_b, out, GG, 64, 1);
}
